// Attention_31765578121660
// MI455X (gfx1250) — compile-verified, matrix operand order checked by reading, not run on hardware
//
#include <hip/hip_runtime.h>
#include <hip/hip_bf16.h>

// ---------------------------------------------------------------------------
// Types / helpers
// ---------------------------------------------------------------------------
typedef __attribute__((ext_vector_type(16))) __bf16 v16bf;
typedef __attribute__((ext_vector_type(2)))  __bf16 v2bf;
typedef __attribute__((ext_vector_type(8)))  float  v8f;

struct __align__(16) U4 { unsigned int x, y, z, w; };
struct __align__(16) F4 { float x, y, z, w; };

union Frag {
    v16bf          v;
    U4             q[2];
    unsigned int   d[8];
    unsigned short u[16];
};

// float -> bf16 via native fptrunc (RNE); backend selects v_cvt_*_bf16_f32
// when the target has it, otherwise expands to the usual bit sequence.
__device__ __forceinline__ unsigned short f2bf(float a) {
    union { __bf16 b; unsigned short u; } cv;
    cv.b = (__bf16)a;
    return cv.u;
}

__device__ __forceinline__ unsigned int f2bf2(float a, float b) {
    union { v2bf v; unsigned int u; } cv;
    cv.v[0] = (__bf16)a;
    cv.v[1] = (__bf16)b;
    return cv.u;
}

__device__ __forceinline__ v8f wmma_bf16(const Frag& a, const Frag& b, v8f c) {
    // (neg_a, A, neg_b, B, c_mod, C, reuse_a, reuse_b)
    return __builtin_amdgcn_wmma_f32_16x16x32_bf16(false, a.v, false, b.v,
                                                   (short)0, c, false, false);
}

// Async copy 16B global -> LDS (ASYNCcnt-tracked, no VGPR round-trip).
// Generic shared pointers: addr[31:0] is the LDS byte offset (flat aperture rule).
__device__ __forceinline__ void async_ld_b128(void* lds_ptr, const void* gptr) {
    unsigned int loff = (unsigned int)(unsigned long long)lds_ptr;
    unsigned long long ga = (unsigned long long)gptr;
    asm volatile("global_load_async_to_lds_b128 %0, %1, off"
                 :: "v"(loff), "v"(ga) : "memory");
}
__device__ __forceinline__ void wait_async0() {
    asm volatile("s_wait_asynccnt 0x0" ::: "memory");
}

// Problem constants (from the reference)
#define BATCH   2
#define NSEQ    2048
#define DIM     1024
#define HEADS   16
#define DHEAD   64
#define INNER   1024           // HEADS * DHEAD
#define QKV_LD  3072           // 3 * INNER
#define MROWS   4096           // BATCH * NSEQ
#define QSCALE  0.125f         // DHEAD^-0.5

// ---------------------------------------------------------------------------
// Kernel 1: QKV projection  C_bf16[4096,3072] = X[4096,1024] @ Wqkv[1024,3072]
// (q columns < 1024 pre-scaled by QSCALE)
// ---------------------------------------------------------------------------
__global__ __launch_bounds__(256) void qkv_gemm_kernel(
    const float* __restrict__ A, const float* __restrict__ B,
    unsigned short* __restrict__ C)
{
    __shared__ __align__(16) unsigned short As[128 * 48];  // [row][k] stride 48
    __shared__ __align__(16) unsigned short Bs[128 * 48];  // [col][k] stride 48 (B transposed)

    const int tid  = threadIdx.x;
    const int wave = tid >> 5;
    const int lane = tid & 31;
    const int ln   = lane & 15;
    const int g    = lane >> 4;
    const int wm   = wave >> 1;      // 0..3  -> 32 rows each
    const int wn   = wave & 1;       // 0..1  -> 64 cols each
    const int bm   = blockIdx.y * 128;
    const int bn   = blockIdx.x * 128;

    v8f acc[2][4];
    for (int ms = 0; ms < 2; ++ms)
        for (int ns = 0; ns < 4; ++ns)
            for (int e = 0; e < 8; ++e) acc[ms][ns][e] = 0.0f;

    for (int k0 = 0; k0 < DIM; k0 += 32) {
        {   // stage A (fp32 -> bf16, packed cvt)
            int row = tid >> 1, ks = (tid & 1) * 16;
            const F4* src = (const F4*)(A + (size_t)(bm + row) * DIM + k0 + ks);
            Frag fa;
            for (int i = 0; i < 4; ++i) {
                F4 v = src[i];
                fa.d[2*i+0] = f2bf2(v.x, v.y);
                fa.d[2*i+1] = f2bf2(v.z, v.w);
            }
            *(U4*)&As[row * 48 + ks]     = fa.q[0];
            *(U4*)&As[row * 48 + ks + 8] = fa.q[1];
        }
        {   // stage B transposed: Bs[n][k] = B[k0+k][bn+n]
            int kk = tid >> 3, ns = (tid & 7) * 16;
            const F4* src = (const F4*)(B + (size_t)(k0 + kk) * QKV_LD + bn + ns);
            for (int i = 0; i < 4; ++i) {
                F4 v = src[i];
                Bs[(ns + 4*i + 0) * 48 + kk] = f2bf(v.x);
                Bs[(ns + 4*i + 1) * 48 + kk] = f2bf(v.y);
                Bs[(ns + 4*i + 2) * 48 + kk] = f2bf(v.z);
                Bs[(ns + 4*i + 3) * 48 + kk] = f2bf(v.w);
            }
        }
        __syncthreads();

        Frag af[2], bf[4];
        for (int ms = 0; ms < 2; ++ms) {
            const unsigned short* ap = &As[(wm * 32 + ms * 16 + ln) * 48];
            af[ms].q[0] = *(const U4*)(ap + g * 8);
            af[ms].q[1] = *(const U4*)(ap + 16 + g * 8);
        }
        for (int ns = 0; ns < 4; ++ns) {
            const unsigned short* bp = &Bs[(wn * 64 + ns * 16 + ln) * 48];
            bf[ns].q[0] = *(const U4*)(bp + g * 8);
            bf[ns].q[1] = *(const U4*)(bp + 16 + g * 8);
        }
        for (int ms = 0; ms < 2; ++ms)
            for (int ns = 0; ns < 4; ++ns)
                acc[ms][ns] = wmma_bf16(af[ms], bf[ns], acc[ms][ns]);
        __syncthreads();
    }

    // epilogue: bf16 store; scale q columns
    for (int ms = 0; ms < 2; ++ms)
        for (int ns = 0; ns < 4; ++ns) {
            int col  = bn + wn * 64 + ns * 16 + ln;
            float sc = (col < INNER) ? QSCALE : 1.0f;
            int rbase = bm + wm * 32 + ms * 16 + 8 * g;
            for (int r = 0; r < 8; ++r)
                C[(size_t)(rbase + r) * QKV_LD + col] = f2bf(acc[ms][ns][r] * sc);
        }
}

// ---------------------------------------------------------------------------
// Kernel 2: causal ALiBi flash attention.
// grid.x = BATCH*HEADS, grid.y = NSEQ/128.  8 waves, 16 q-rows per wave.
// ---------------------------------------------------------------------------
__global__ __launch_bounds__(256) void attn_kernel(
    const unsigned short* __restrict__ qkv,   // bf16 [4096, 3072]
    unsigned short* __restrict__ attn_out)    // bf16 [4096, 1024]
{
    __shared__ __align__(16) unsigned short Ks[64 * 72];        // [key][d]
    __shared__ __align__(16) unsigned short Vs[64 * 72];        // [d][key] (transposed)
    __shared__ __align__(16) unsigned short Ps[8 * 16 * 72];    // per-wave P tiles

    const int tid  = threadIdx.x;
    const int wave = tid >> 5;
    const int lane = tid & 31;
    const int ln   = lane & 15;
    const int g    = lane >> 4;
    const int bh   = blockIdx.x;
    const int bb   = bh >> 4;
    const int h    = bh & 15;
    const int qbase = blockIdx.y * 128;
    const float slope = exp2f(-0.5f * (float)(h + 1));   // ALiBi slope for 16 heads

    const size_t rowbase = (size_t)bb * NSEQ;

    // Q fragments: rows qbase + wave*16 + ln, d = 0..63 (already scaled in k1)
    Frag qf[2];
    {
        const unsigned short* qp =
            qkv + (rowbase + qbase + wave * 16 + ln) * (size_t)QKV_LD + h * DHEAD;
        for (int f = 0; f < 2; ++f) {
            qf[f].q[0] = *(const U4*)(qp + f * 32 + g * 8);
            qf[f].q[1] = *(const U4*)(qp + f * 32 + 16 + g * 8);
        }
    }

    float m[8], lsum[8];
    v8f o[4];
    for (int r = 0; r < 8; ++r) { m[r] = -1e30f; lsum[r] = 0.0f; }
    for (int t = 0; t < 4; ++t)
        for (int e = 0; e < 8; ++e) o[t][e] = 0.0f;

    const int jend = qbase + 128;                 // causal: keys beyond q block never needed
    for (int j0 = 0; j0 < jend; j0 += 64) {
        {   // stage K via async-to-LDS; V transposed manually
            int key = tid >> 2;
            int ds  = (tid & 3) * 16;
            size_t base = (rowbase + j0 + key) * (size_t)QKV_LD + h * DHEAD + ds;
            // K tile: straight row-major copy -> async path
            async_ld_b128(&Ks[key * 72 + ds],     qkv + base + INNER);
            async_ld_b128(&Ks[key * 72 + ds + 8], qkv + base + INNER + 8);
            // V tile: transpose during staging (layout change, VGPR path)
            const U4* vsrc = (const U4*)(qkv + base + 2 * INNER);
            Frag vv;
            vv.q[0] = vsrc[0]; vv.q[1] = vsrc[1];
            for (int i = 0; i < 16; ++i) Vs[(ds + i) * 72 + key] = vv.u[i];
        }
        wait_async0();
        __syncthreads();

        // --- S = Q K^T (+ ALiBi, causal mask) ---
        v8f s[4];
        float tmax[8];
        for (int r = 0; r < 8; ++r) tmax[r] = -1e30f;
        const int i_base = qbase + wave * 16 + 8 * g;
        for (int kt = 0; kt < 4; ++kt) {
            Frag kf0, kf1;
            const unsigned short* kp = &Ks[(kt * 16 + ln) * 72];
            kf0.q[0] = *(const U4*)(kp + g * 8);
            kf0.q[1] = *(const U4*)(kp + 16 + g * 8);
            kf1.q[0] = *(const U4*)(kp + 32 + g * 8);
            kf1.q[1] = *(const U4*)(kp + 48 + g * 8);
            v8f a;
            for (int e = 0; e < 8; ++e) a[e] = 0.0f;
            a = wmma_bf16(qf[0], kf0, a);
            a = wmma_bf16(qf[1], kf1, a);
            const int jj = j0 + kt * 16 + ln;
            for (int r = 0; r < 8; ++r) {
                int irow = i_base + r;
                float sv = a[r];
                sv = (jj > irow) ? -1e30f : (sv - (float)(irow - jj) * slope);
                a[r] = sv;
                tmax[r] = fmaxf(tmax[r], sv);
            }
            s[kt] = a;
        }
        // row-max across the 16-lane group
        for (int r = 0; r < 8; ++r) {
            float v = tmax[r];
            for (int off = 1; off < 16; off <<= 1)
                v = fmaxf(v, __shfl_xor(v, off, 32));
            tmax[r] = v;
        }
        // online softmax rescale
        float fac[8];
        for (int r = 0; r < 8; ++r) {
            float mn = fmaxf(m[r], tmax[r]);
            fac[r] = __expf(m[r] - mn);
            m[r] = mn;
            lsum[r] *= fac[r];
        }
        for (int t = 0; t < 4; ++t)
            for (int r = 0; r < 8; ++r) o[t][r] *= fac[r];

        // P = exp(S - m) -> bf16 via per-wave LDS (C-layout -> A-layout transpose)
        float rsum[8];
        for (int r = 0; r < 8; ++r) rsum[r] = 0.0f;
        unsigned short* pp = &Ps[wave * 16 * 72];
        for (int kt = 0; kt < 4; ++kt)
            for (int r = 0; r < 8; ++r) {
                float p = __expf(s[kt][r] - m[r]);
                rsum[r] += p;
                pp[(8 * g + r) * 72 + kt * 16 + ln] = f2bf(p);
            }
        for (int r = 0; r < 8; ++r) {
            float v = rsum[r];
            for (int off = 1; off < 16; off <<= 1)
                v += __shfl_xor(v, off, 32);
            lsum[r] += v;
        }
        // wave-private LDS region: wait on own DS ops (in-order per wave)
        asm volatile("s_wait_dscnt 0x0" ::: "memory");

        // --- O += P V ---
        Frag pf0, pf1;
        {
            const unsigned short* pr = pp + ln * 72;
            pf0.q[0] = *(const U4*)(pr + g * 8);
            pf0.q[1] = *(const U4*)(pr + 16 + g * 8);
            pf1.q[0] = *(const U4*)(pr + 32 + g * 8);
            pf1.q[1] = *(const U4*)(pr + 48 + g * 8);
        }
        for (int t = 0; t < 4; ++t) {
            Frag vf0, vf1;
            const unsigned short* vp = &Vs[(t * 16 + ln) * 72];
            vf0.q[0] = *(const U4*)(vp + g * 8);
            vf0.q[1] = *(const U4*)(vp + 16 + g * 8);
            vf1.q[0] = *(const U4*)(vp + 32 + g * 8);
            vf1.q[1] = *(const U4*)(vp + 48 + g * 8);
            o[t] = wmma_bf16(pf0, vf0, o[t]);
            o[t] = wmma_bf16(pf1, vf1, o[t]);
        }
        __syncthreads();
    }

    // epilogue: normalize (precomputed reciprocal) and store bf16 in [b,n,h*d]
    float rinv[8];
    for (int r = 0; r < 8; ++r) rinv[r] = 1.0f / lsum[r];
    for (int t = 0; t < 4; ++t)
        for (int r = 0; r < 8; ++r) {
            int irow = qbase + wave * 16 + 8 * g + r;
            attn_out[(rowbase + irow) * (size_t)INNER + h * DHEAD + t * 16 + ln] =
                f2bf(o[t][r] * rinv[r]);
        }
}

// ---------------------------------------------------------------------------
// Kernel 3: output projection  out[4096,1024] = attn_bf16 @ Wout + b_out
// ---------------------------------------------------------------------------
__global__ __launch_bounds__(256) void out_gemm_kernel(
    const unsigned short* __restrict__ A,   // bf16 [4096, 1024]
    const float* __restrict__ B,            // [1024, 1024]
    const float* __restrict__ bias,         // [1024]
    float* __restrict__ C)                  // [4096, 1024]
{
    __shared__ __align__(16) unsigned short As[128 * 48];
    __shared__ __align__(16) unsigned short Bs[128 * 48];

    const int tid  = threadIdx.x;
    const int wave = tid >> 5;
    const int lane = tid & 31;
    const int ln   = lane & 15;
    const int g    = lane >> 4;
    const int wm   = wave >> 1;
    const int wn   = wave & 1;
    const int bm   = blockIdx.y * 128;
    const int bn   = blockIdx.x * 128;

    v8f acc[2][4];
    for (int ms = 0; ms < 2; ++ms)
        for (int ns = 0; ns < 4; ++ns)
            for (int e = 0; e < 8; ++e) acc[ms][ns][e] = 0.0f;

    for (int k0 = 0; k0 < INNER; k0 += 32) {
        {   // stage A (bf16 straight copy -> async-to-LDS path)
            int row = tid >> 1, ks = (tid & 1) * 16;
            const unsigned short* src = A + (size_t)(bm + row) * INNER + k0 + ks;
            async_ld_b128(&As[row * 48 + ks],     src);
            async_ld_b128(&As[row * 48 + ks + 8], src + 8);
        }
        {   // stage B transposed (fp32 -> bf16)
            int kk = tid >> 3, ns = (tid & 7) * 16;
            const F4* src = (const F4*)(B + (size_t)(k0 + kk) * DIM + bn + ns);
            for (int i = 0; i < 4; ++i) {
                F4 v = src[i];
                Bs[(ns + 4*i + 0) * 48 + kk] = f2bf(v.x);
                Bs[(ns + 4*i + 1) * 48 + kk] = f2bf(v.y);
                Bs[(ns + 4*i + 2) * 48 + kk] = f2bf(v.z);
                Bs[(ns + 4*i + 3) * 48 + kk] = f2bf(v.w);
            }
        }
        wait_async0();
        __syncthreads();

        Frag af[2], bf[4];
        for (int ms = 0; ms < 2; ++ms) {
            const unsigned short* ap = &As[(wm * 32 + ms * 16 + ln) * 48];
            af[ms].q[0] = *(const U4*)(ap + g * 8);
            af[ms].q[1] = *(const U4*)(ap + 16 + g * 8);
        }
        for (int ns = 0; ns < 4; ++ns) {
            const unsigned short* bp = &Bs[(wn * 64 + ns * 16 + ln) * 48];
            bf[ns].q[0] = *(const U4*)(bp + g * 8);
            bf[ns].q[1] = *(const U4*)(bp + 16 + g * 8);
        }
        for (int ms = 0; ms < 2; ++ms)
            for (int ns = 0; ns < 4; ++ns)
                acc[ms][ns] = wmma_bf16(af[ms], bf[ns], acc[ms][ns]);
        __syncthreads();
    }

    for (int ms = 0; ms < 2; ++ms)
        for (int ns = 0; ns < 4; ++ns) {
            int col = bn + wn * 64 + ns * 16 + ln;
            float bv = bias[col];
            int rbase = bm + wm * 32 + ms * 16 + 8 * g;
            for (int r = 0; r < 8; ++r)
                C[(size_t)(rbase + r) * DIM + col] = acc[ms][ns][r] + bv;
        }
}

// ---------------------------------------------------------------------------
// Host entry
// ---------------------------------------------------------------------------
extern "C" void kernel_launch(void* const* d_in, const int* in_sizes, int n_in,
                              void* d_out, int out_size, void* d_ws, size_t ws_size,
                              hipStream_t stream)
{
    (void)in_sizes; (void)n_in; (void)out_size; (void)ws_size;
    const float* x     = (const float*)d_in[0];
    const float* w_qkv = (const float*)d_in[1];
    const float* w_out = (const float*)d_in[2];
    const float* b_out = (const float*)d_in[3];
    float* out = (float*)d_out;

    unsigned short* qkv_bf16  = (unsigned short*)d_ws;                   // 4096*3072 bf16 (24 MB)
    unsigned short* attn_bf16 = qkv_bf16 + (size_t)MROWS * QKV_LD;       // 4096*1024 bf16 (8 MB)

    qkv_gemm_kernel<<<dim3(QKV_LD / 128, MROWS / 128), 256, 0, stream>>>(x, w_qkv, qkv_bf16);
    attn_kernel<<<dim3(BATCH * HEADS, NSEQ / 128), 256, 0, stream>>>(qkv_bf16, attn_bf16);
    out_gemm_kernel<<<dim3(DIM / 128, MROWS / 128), 256, 0, stream>>>(attn_bf16, w_out, b_out, out);
}
